// OptimizedKANConv2D_49976239456990
// MI455X (gfx1250) — matrix operand order repaired and checked by reading, not compiled, not run on hardware
//
#include <hip/hip_runtime.h>
#include <cstdint>

// ---- problem constants (match reference) ----
#define IN_C   32
#define OUT_C  64
#define HW     56
#define NB     4
#define NPOS   (NB*HW*HW)        // 12544 output positions
#define PATCH  (IN_C*3*3)        // 288
#define NF     8                 // bases per patch element (grid_size+order)
#define KDIM   (PATCH*NF)        // 2304 contraction length
#define TM     64                // spatial rows per block
#define PCH    32                // patch elements per K-step
#define KCH    (PCH*NF)          // 256 spline-K per step
#define NSTEP  (PATCH/PCH)       // 9

typedef _Float16 h8   __attribute__((ext_vector_type(8)));
typedef _Float16 v16h __attribute__((ext_vector_type(16)));
typedef float    v8f  __attribute__((ext_vector_type(8)));

// Convert fp32 weights to f16 in workspace (L2-resident, reused by all blocks).
__global__ __launch_bounds__(256) void kan_prep(const float* __restrict__ bw,
                                                const float* __restrict__ sw,
                                                _Float16* __restrict__ swh,
                                                _Float16* __restrict__ bwh)
{
    const int i = blockIdx.x * blockDim.x + threadIdx.x;
    if (i < OUT_C * KDIM)  swh[i] = (_Float16)sw[i];
    if (i < OUT_C * PATCH) bwh[i] = (_Float16)bw[i];
}

__device__ __forceinline__ v16h cat16(h8 lo, h8 hi) {
    return __builtin_shufflevector(lo, hi, 0,1,2,3,4,5,6,7,8,9,10,11,12,13,14,15);
}

__global__ __launch_bounds__(128) void kan_main(const float*    __restrict__ x,
                                                const _Float16* __restrict__ swh,
                                                const _Float16* __restrict__ bwh,
                                                const float*    __restrict__ scaler,
                                                float*          __restrict__ out)
{
    // Double-buffered tiles: one barrier per K-step, fill(s+1) overlaps wmma(s).
    __shared__ __attribute__((aligned(16))) _Float16 As[2][TM][KCH]; // bases (B layout), 64 KB
    __shared__ __attribute__((aligned(16))) _Float16 Ab[2][TM][PCH]; // raw x patches,    8 KB
    __shared__ __attribute__((aligned(16))) _Float16 trash[64];      // OOB scatter sink

    const int tid   = threadIdx.x;
    const int lane  = tid & 31;
    const int wave  = tid >> 5;          // 0..3 -> channel tile; also fill row group
    const int hi    = lane >> 4;         // fragment half selector
    const int l16   = lane & 15;
    const int nbase = blockIdx.x * TM;

    v8f accS[4] = {};                    // spline accumulators (4 spatial col tiles)
    v8f accB[4] = {};                    // base-conv accumulators

    const float t0 = -2.2f;              // first knot; h = 0.4, 1/h = 2.5
    _Float16* const sink = &trash[lane]; // per-lane dump slot (spread across banks)

    // ---- step-invariant per-thread fill coordinates (pl = lane, nl = wave+4i) ----
    int pbase[16], oyA[16], oxA[16];
    #pragma unroll
    for (int i = 0; i < 16; ++i) {
        const int n   = nbase + wave + 4*i;
        const int b   = n / (HW*HW);
        const int rem = n - b*(HW*HW);
        const int oy  = rem / HW;
        const int ox  = rem - oy*HW;
        pbase[i] = (b*IN_C*HW + oy)*HW + ox;
        oyA[i]   = oy;
        oxA[i]   = ox;
    }

    // ---- fill: gather x, closed-form uniform cubic bases; fully branch-free ----
    auto fill = [&](int step, int buf) {
        const int p    = step*PCH + lane;    // this thread's patch element
        const int c    = p / 9;
        const int r9   = p - c*9;
        const int kh   = r9 / 3;
        const int kw   = r9 - kh*3;
        const int coff = c*(HW*HW) + (kh-1)*HW + (kw-1);
        #pragma unroll
        for (int i = 0; i < 16; ++i) {
            const int  nl  = wave + 4*i;
            const int  iy  = oyA[i] + kh - 1;
            const int  ix  = oxA[i] + kw - 1;
            const bool inb = ((unsigned)iy < (unsigned)HW) & ((unsigned)ix < (unsigned)HW);
            const int  adr = inb ? (pbase[i] + coff) : 0;   // safe address
            float xv = x[adr];
            xv = inb ? xv : 0.0f;        // zero padding feeds the spline (as in reference)
            Ab[buf][nl][lane] = (_Float16)xv;

            _Float16* dst = &As[buf][nl][lane*NF];
            const h8 zz = {};
            *(h8*)dst = zz;                               // clear all 8 bases

            // x in [t_j, t_j+h): 4 non-zero uniform cubic bases f = j-3..j.
            const float rel = (xv - t0) * 2.5f;           // (x - t0)/h
            const float fj  = floorf(rel);
            const int   j   = (int)fj;
            const float u   = rel - fj;
            const float um  = 1.0f - u;
            const float u2  = u*u, u3 = u2*u;
            float w[4];
            w[0] = um*um*um * (1.0f/6.0f);                // f = j-3
            w[1] = 0.5f*u3 - u2 + (2.0f/3.0f);            // f = j-2
            w[3] = u3 * (1.0f/6.0f);                      // f = j
            w[2] = 1.0f - w[0] - w[1] - w[3];             // f = j-1 (partition of unity)
            #pragma unroll
            for (int k = 0; k < 4; ++k) {
                const int f = j - 3 + k;
                // out-of-span bases (incl. x outside the knot vector) go to the sink
                _Float16* q = ((unsigned)f < (unsigned)NF) ? (dst + f) : sink;
                *q = (_Float16)w[k];
            }
        }
    };

    // ---- wmma: 4 base + 32 spline WMMAs per wave per step ----
    auto mm = [&](int step, int buf) {
        const int p0   = step * PCH;
        const int orow = wave*16 + l16;                   // channel row (A matrix M)
        {
            const h8* wp = (const h8*)(bwh + orow*PATCH + p0);
            const v16h afrag = cat16(wp[2*hi], wp[2*hi + 1]);
            #pragma unroll
            for (int ct = 0; ct < 4; ++ct) {
                const h8* bp = (const h8*)(&Ab[buf][ct*16 + l16][16*hi]);
                accB[ct] = __builtin_amdgcn_wmma_f32_16x16x32_f16(
                    false, afrag, false, cat16(bp[0], bp[1]), (short)0,
                    accB[ct], false, false);
            }
        }
        #pragma unroll
        for (int kc = 0; kc < 8; ++kc) {
            const h8* wp = (const h8*)(swh + orow*KDIM + p0*NF + kc*32);
            const v16h afrag = cat16(wp[2*hi], wp[2*hi + 1]);
            #pragma unroll
            for (int ct = 0; ct < 4; ++ct) {
                const h8* bp = (const h8*)(&As[buf][ct*16 + l16][kc*32 + 16*hi]);
                accS[ct] = __builtin_amdgcn_wmma_f32_16x16x32_f16(
                    false, afrag, false, cat16(bp[0], bp[1]), (short)0,
                    accS[ct], false, false);
            }
        }
    };

    // ---- software pipeline: one barrier per step ----
    fill(0, 0);
    __syncthreads();
    for (int s = 0; s < NSTEP; ++s) {
        if (s + 1 < NSTEP) fill(s + 1, (s + 1) & 1);  // writes buf (s+1)&1
        mm(s, s & 1);                                  // reads  buf  s&1
        __syncthreads();
    }

    // ---- epilogue: out = SiLU(base) + scaler[o]*spline ; coalesced along x ----
    #pragma unroll
    for (int ct = 0; ct < 4; ++ct) {
        const int n   = nbase + ct*16 + l16;
        const int b   = n / (HW*HW);
        const int rem = n - b*(HW*HW);
        const int oy  = rem / HW;
        const int ox  = rem - oy*HW;
        #pragma unroll
        for (int v = 0; v < 8; ++v) {
            const int o  = wave*16 + v + 8*hi;        // D layout: M = v + 8*(lane>=16)
            const float bb = accB[ct][v];
            const float si = bb / (1.0f + __expf(-bb));
            out[((b*OUT_C + o)*HW + oy)*HW + ox] = si + scaler[o]*accS[ct][v];
        }
    }
}

extern "C" void kernel_launch(void* const* d_in, const int* in_sizes, int n_in,
                              void* d_out, int out_size, void* d_ws, size_t ws_size,
                              hipStream_t stream)
{
    const float* x  = (const float*)d_in[0];   // (4,32,56,56)
    const float* bw = (const float*)d_in[1];   // (64,32,3,3)
    const float* sw = (const float*)d_in[2];   // (64,288,8)
    const float* sc = (const float*)d_in[3];   // (64,)
    float* out = (float*)d_out;                // (4,64,56,56)

    _Float16* swh = (_Float16*)d_ws;           // 64*2304 halves
    _Float16* bwh = swh + OUT_C * KDIM;        // 64*288 halves (total ws ~324 KB)

    const int total = OUT_C * KDIM;
    kan_prep<<<(total + 255) / 256, 256, 0, stream>>>(bw, sw, swh, bwh);
    kan_main<<<NPOS / TM, 128, 0, stream>>>(x, swh, bwh, sc, out);

    (void)in_sizes; (void)n_in; (void)out_size; (void)ws_size;
}
